// CostVolumeLayer_13417477833614
// MI455X (gfx1250) — matrix operand order repaired and checked by reading, not compiled, not run on hardware
//
#include <hip/hip_runtime.h>

// Cost volume via banded bf16 WMMA on gfx1250 (wave32).
// B=8, H=128, W=256, C=128, r=4 -> 81 offsets.
// Block = 8 waves; wave i owns output row h0+i, 32 w positions (2 M-tiles).
// Warp rows staged one at a time (double-buffered bf16 strip in LDS) and
// shared by all 9 output rows that need them -> ~2.5x warp traffic vs 10x.

typedef __attribute__((ext_vector_type(16))) __bf16 v16bf;
typedef __attribute__((ext_vector_type(8)))  float  v8f;

#define BB 8
#define HH 128
#define WW 256
#define CC 128
#define RR 4
#define MAXOFF 9              // 2*RR+1
#define TH 8                  // output rows per block (one per wave)
#define TW 32                 // output w positions per block
#define NPOS (TW + 2*RR)      // 40 staged warp positions per row
#define NG (TH + 2*RR)        // 16 warp rows swept per block

union FragA { v16bf v; unsigned int u[8]; };
union FragB { v16bf v; uint4 q[2]; };

__device__ __forceinline__ unsigned int packbf(float lo, float hi) {
    unsigned int a = __float_as_uint(lo);
    unsigned int b = __float_as_uint(hi);
    a += 0x7FFFu + ((a >> 16) & 1u);      // round-to-nearest-even bf16
    b += 0x7FFFu + ((b >> 16) & 1u);
    return (a >> 16) | (b & 0xFFFF0000u);
}

// Stage one warp row (40 positions x 128 ch) into an LDS strip as bf16.
__device__ __forceinline__ void stage_row(__bf16* dst, const float* __restrict__ warp,
                                          int bIdx, int g, int w0base, int tid) {
    #pragma unroll
    for (int it = 0; it < (NPOS * (CC / 4)) / 256; ++it) {   // 5 iters
        const int idx = tid + it * 256;
        const int c4  = idx & 31;        // channel group of 4
        const int pos = idx >> 5;        // 0..39
        const int gw  = w0base - RR + pos;
        unsigned int p0 = 0u, p1 = 0u;   // zero padding
        if (g >= 0 && g < HH && gw >= 0 && gw < WW) {
            const float4 f = *(const float4*)(
                warp + ((bIdx * HH + g) * WW + gw) * CC + c4 * 4);
            p0 = packbf(f.x, f.y);
            p1 = packbf(f.z, f.w);
        }
        *(uint2*)(dst + pos * CC + c4 * 4) = make_uint2(p0, p1);
    }
}

__global__ __launch_bounds__(256) void cost_volume_wmma(
    const float* __restrict__ c1, const float* __restrict__ warp,
    float* __restrict__ out)
{
    __shared__ __align__(16) __bf16 s_warp[2][NPOS * CC];   // 2 x 10 KB = 20 KB

    const int tid   = threadIdx.x;
    const int wavei = tid >> 5;          // 0..7 -> output row offset
    const int lane  = tid & 31;
    const int n     = lane & 15;         // A row / B col / D col
    const int half  = lane >> 4;         // K half-select per ISA layout

    int wg = blockIdx.x;                 // 8 * 16 * 8 = 1024 blocks
    const int wgw = wg & 7;              // W / TW = 8
    wg >>= 3;
    const int hblk = wg & 15;            // H / TH = 16
    const int bIdx = wg >> 4;

    const int w0base = wgw * TW;
    const int h0     = hblk * TH;
    const int h      = h0 + wavei;

    // ---- A fragments: c1[b,h,w0+j*16+n,:] -> bf16, 2 w-tiles x 4 k-steps ----
    // ISA 16-bit A 16x32 layout: elems 0..7 = K k0+8*half.., elems 8..15 = K k0+16+8*half..
    FragA afrag[2][4];
    #pragma unroll
    for (int j = 0; j < 2; ++j) {
        const float* ap = c1 + ((bIdx * HH + h) * WW + (w0base + j * 16 + n)) * CC;
        #pragma unroll
        for (int s = 0; s < 4; ++s) {
            const int k0 = s * 32;
            const float4 a0 = *(const float4*)(ap + k0 + half * 8);
            const float4 a1 = *(const float4*)(ap + k0 + half * 8 + 4);
            const float4 b0 = *(const float4*)(ap + k0 + 16 + half * 8);
            const float4 b1 = *(const float4*)(ap + k0 + 16 + half * 8 + 4);
            afrag[j][s].u[0] = packbf(a0.x, a0.y);
            afrag[j][s].u[1] = packbf(a0.z, a0.w);
            afrag[j][s].u[2] = packbf(a1.x, a1.y);
            afrag[j][s].u[3] = packbf(a1.z, a1.w);
            afrag[j][s].u[4] = packbf(b0.x, b0.y);
            afrag[j][s].u[5] = packbf(b0.z, b0.w);
            afrag[j][s].u[6] = packbf(b1.x, b1.y);
            afrag[j][s].u[7] = packbf(b1.z, b1.w);
        }
    }

    const int obase_row = (bIdx * HH + h) * WW;   // row base in out pixels

    // Prologue: stage first warp row (g = h0-4) into buffer 0.
    stage_row(s_warp[0], warp, bIdx, h0 - RR, w0base, tid);
    __syncthreads();

    for (int gi = 0; gi < NG; ++gi) {
        // Stage next row into the other buffer while computing this one.
        if (gi + 1 < NG)
            stage_row(s_warp[(gi + 1) & 1], warp, bIdx, h0 - RR + gi + 1, w0base, tid);
        if (gi + 2 < NG) {
            const int g2 = h0 - RR + gi + 2;
            if (g2 >= 0 && g2 < HH)                 // global_prefetch of row g+2
                __builtin_prefetch(warp + ((bIdx * HH + g2) * WW + w0base) * CC + tid * 16,
                                   0, 1);
        }

        const int dy = gi - wavei;                  // g - h + RR
        if (dy >= 0 && dy <= 2 * RR) {              // wave-uniform: EXEC all 1s inside
            const __bf16* buf = s_warp[gi & 1];
            #pragma unroll
            for (int j = 0; j < 2; ++j) {
                #pragma unroll
                for (int nt = 0; nt < 2; ++nt) {
                    // B 32x16: lane n = column (warp position), elems 0..7 = K k0+8*half.., elems 8..15 = K k0+16+8*half..
                    int posidx = j * 16 + nt * 16 + n;
                    if (posidx > NPOS - 1) posidx = NPOS - 1;   // clamp unused cols

                    v8f acc = {};
                    #pragma unroll
                    for (int s = 0; s < 4; ++s) {
                        FragB bfrag;
                        const __bf16* bp = buf + posidx * CC + s * 32 + half * 8;
                        bfrag.q[0] = *(const uint4*)(bp);
                        bfrag.q[1] = *(const uint4*)(bp + 16);
                        acc = __builtin_amdgcn_wmma_f32_16x16x32_bf16(
                            false, afrag[j][s].v, false, bfrag.v,
                            (short)0, acc, false, false);
                    }

                    // D: row mrow = v + 8*half, col n.  dx+4 = nt*16 + n - mrow.
                    // Along v the out index moves by +81-1 = +80 elems -> one base
                    // address, compile-time immediate offsets for all 8 stores.
                    const int dx0 = nt * 16 + n - 8 * half;     // dx+4 at v=0
                    float* op = out + (obase_row + (w0base + j * 16 + 8 * half))
                                       * (MAXOFF * MAXOFF)
                                    + dy * MAXOFF + dx0;
                    #pragma unroll
                    for (int v = 0; v < 8; ++v) {
                        if (dx0 - v >= 0 && dx0 - v <= 2 * RR) {
                            float val = acc[v] * (1.0f / 128.0f);   // mean over C
                            val = (val >= 0.0f) ? val : 0.1f * val; // leaky relu
                            op[v * 80] = val;
                        }
                    }
                }
            }
        }
        __syncthreads();   // buf[gi&1] free to restage, buf[(gi+1)&1] ready
    }
}

extern "C" void kernel_launch(void* const* d_in, const int* in_sizes, int n_in,
                              void* d_out, int out_size, void* d_ws, size_t ws_size,
                              hipStream_t stream) {
    const float* c1   = (const float*)d_in[0];
    const float* warp = (const float*)d_in[1];
    float* out = (float*)d_out;
    (void)in_sizes; (void)n_in; (void)out_size; (void)d_ws; (void)ws_size;

    dim3 grid(BB * (HH / TH) * (WW / TW));   // 8*16*8 = 1024
    dim3 block(256);                         // 8 waves of 32
    hipLaunchKernelGGL(cost_volume_wmma, grid, block, 0, stream, c1, warp, out);
}
